// LASAS_8108898255194
// MI455X (gfx1250) — matrix operand order repaired and checked by reading, not compiled, not run on hardware
//
#include <hip/hip_runtime.h>

// ---------------------------------------------------------------------------
// Problem constants (from reference): B=8, T=2048 -> 16384 rows, DIM=512,
// N_HEAD=8, OUT_DIM=512, D_K=64 (scale 1/8), Wtd rows = 504.
// ---------------------------------------------------------------------------
#define NROWS   16384
#define DIM     512
#define NHEAD   8
#define TDROWS  504

typedef __attribute__((ext_vector_type(16))) __bf16 v16bf;
typedef __attribute__((ext_vector_type(8)))  __bf16 v8bf;
typedef __attribute__((ext_vector_type(8)))  float  v8f;

// ===========================================================================
// Prep 1: u[h] = Wt[h]^T ba[h], w[h] = Wa[h]^T bt[h], c[h] = ba.bt
// ===========================================================================
__global__ void k_prep_uwc(const float* __restrict__ Wa, const float* __restrict__ ba,
                           const float* __restrict__ Wt, const float* __restrict__ bt,
                           float* __restrict__ u, float* __restrict__ w,
                           float* __restrict__ c)
{
    const int h = blockIdx.x;      // 8 blocks
    const int j = threadIdx.x;     // 512 threads
    float ua = 0.f, wv = 0.f;
    for (int o = 0; o < DIM; ++o) {
        ua += ba[h*DIM + o] * Wt[((size_t)h*DIM + o)*DIM + j];
        wv += bt[h*DIM + o] * Wa[((size_t)h*DIM + o)*DIM + j];
    }
    u[h*DIM + j] = ua;
    w[h*DIM + j] = wv;
    if (j == 0) {
        float cc = 0.f;
        for (int o = 0; o < DIM; ++o) cc += ba[h*DIM + o] * bt[h*DIM + o];
        c[h] = cc;
    }
}

// ===========================================================================
// Prep 2: M[h] = Wa[h]^T Wt[h], stored bf16 in WMMA-B-fragment swizzled
// layout:  blocks indexed (h, nt, kt); per block 32 lanes x 16 halves where
// lane l, elem e  <->  Bmat[j = kt*32 + (l<16?0:16)+e][i = nt*16 + l%16],
// Bmat[j][i] = sum_o Wa[o,i] * Wt[o,j].
// ===========================================================================
__global__ void k_prep_M(const float* __restrict__ Wa, const float* __restrict__ Wt,
                         __bf16* __restrict__ Bswz)
{
    __shared__ float bounce[8][32*16];               // per-wave 32(j) x 16(i)
    const int tid  = threadIdx.x;
    const int wave = tid >> 5;
    const int lane = tid & 31;
    const int lrow = lane & 15;
    const int g  = blockIdx.x*8 + wave;              // 0..4095
    const int h  = g >> 9;                           // /512
    const int kt = (g >> 5) & 15;                    // 16 K-tiles (j, 32 each)
    const int nt = g & 31;                           // 32 N-tiles (i, 16 each)
    const int j0 = kt*32 + lrow;
    const int i  = nt*16 + lrow;
    const float* WtH = Wt + (size_t)h*DIM*DIM;
    const float* WaH = Wa + (size_t)h*DIM*DIM;

    v8f c0, c1;
#pragma unroll
    for (int v = 0; v < 8; ++v) { c0[v] = 0.f; c1[v] = 0.f; }

    for (int ks = 0; ks < 16; ++ks) {                // K = o, 32 per step
        const int ob = ks*32;
        v16bf a0f, a1f, bfrag;
#pragma unroll
        for (int e = 0; e < 16; ++e) {
            int oa = ob + ((lane < 16) ? 0 : 16) + e;
            a0f[e] = (__bf16)WtH[oa*DIM + j0];
            a1f[e] = (__bf16)WtH[oa*DIM + j0 + 16];
            int obb = ob + ((lane < 16) ? 0 : 16) + e;
            bfrag[e] = (__bf16)WaH[obb*DIM + i];
        }
        c0 = __builtin_amdgcn_wmma_f32_16x16x32_bf16(false, a0f, false, bfrag,
                                                     (short)0, c0, false, false);
        c1 = __builtin_amdgcn_wmma_f32_16x16x32_bf16(false, a1f, false, bfrag,
                                                     (short)0, c1, false, false);
    }
    // Bounce C tiles through LDS to reshape into B-fragment layout.
    float* ld = bounce[wave];
#pragma unroll
    for (int v = 0; v < 8; ++v) {
        ld[(      v + 8*(lane >> 4))*16 + lrow] = c0[v];   // j rows 0..15
        ld[(16 +  v + 8*(lane >> 4))*16 + lrow] = c1[v];   // j rows 16..31
    }
    __syncthreads();
    v16bf outv;
#pragma unroll
    for (int e = 0; e < 16; ++e) {
        int jl = ((lane < 16) ? 0 : 16) + e;
        outv[e] = (__bf16)ld[jl*16 + lrow];
    }
    *(v16bf*)(Bswz + (size_t)((h*32 + nt)*16 + kt)*512 + lane*16) = outv;
}

// ===========================================================================
// Prep 3: Wtd -> bf16, padded 504->512 cols, swizzled B-fragment layout.
// ===========================================================================
__global__ void k_prep_wtd(const float* __restrict__ Wtd, __bf16* __restrict__ Wswz)
{
    const int id   = blockIdx.x*256 + threadIdx.x;   // 0..16383
    const int nt   = id >> 9;                        // 0..31
    const int kt   = (id >> 5) & 15;                 // 0..15
    const int lane = id & 31;
    const int lrow = lane & 15;
    const int o    = nt*16 + lrow;
    v16bf v;
#pragma unroll
    for (int e = 0; e < 16; ++e) {
        int j = kt*32 + ((lane < 16) ? 0 : 16) + e;
        v[e] = (o < TDROWS) ? (__bf16)Wtd[(size_t)o*DIM + j] : (__bf16)0.f;
    }
    *(v16bf*)(Wswz + (size_t)(nt*16 + kt)*512 + lane*16) = v;
}

// ===========================================================================
// Prep 4: tb[r][h] = (u[h].x_t[r] + c[h]) / 8   (one wave per row)
// ===========================================================================
__global__ void k_prep_tb(const float* __restrict__ text, const float* __restrict__ u,
                          const float* __restrict__ c, float* __restrict__ tb)
{
    const int wave = threadIdx.x >> 5;
    const int lane = threadIdx.x & 31;
    const int r = blockIdx.x*8 + wave;
    float acc[NHEAD];
#pragma unroll
    for (int h = 0; h < NHEAD; ++h) acc[h] = 0.f;
    for (int it = 0; it < 16; ++it) {
        const int j = lane + it*32;
        const float xv = text[(size_t)r*DIM + j];
#pragma unroll
        for (int h = 0; h < NHEAD; ++h) acc[h] += xv * u[h*DIM + j];
    }
#pragma unroll
    for (int h = 0; h < NHEAD; ++h) {
        acc[h] += __shfl_xor(acc[h], 16, 32);
        acc[h] += __shfl_xor(acc[h],  8, 32);
        acc[h] += __shfl_xor(acc[h],  4, 32);
        acc[h] += __shfl_xor(acc[h],  2, 32);
        acc[h] += __shfl_xor(acc[h],  1, 32);
    }
    if (lane == 0) {
#pragma unroll
        for (int h = 0; h < NHEAD; ++h)
            tb[(size_t)r*NHEAD + h] = (acc[h] + c[h]) * 0.125f;
    }
}

// ===========================================================================
// Main fused kernel. 64 rows per WG (256 threads = 8 waves).
// Xt staged to LDS as bf16 (64 KB); each wave then hoists its 16 A-fragments
// (full K=512 of its 16-row subtile) into registers (128 VGPRs) and streams
// the L2-resident swizzled B blocks: inner loop = 1 global b256 load + wmma.
// Xa is read directly from global f32 in the epilogue (coalesced 64B/half-wave).
// ===========================================================================
__global__ __launch_bounds__(256)
void k_main(const float* __restrict__ acoustic, const float* __restrict__ text,
            const __bf16* __restrict__ Bswz, const __bf16* __restrict__ Wtdswz,
            const float* __restrict__ wvec, const float* __restrict__ btd,
            const float* __restrict__ tb, float* __restrict__ out)
{
    extern __shared__ char dsm[];
    __bf16* xt = (__bf16*)dsm;            // [64][512] bf16 = 64 KB
    const int tid = threadIdx.x;
    const long rowbase = (long)blockIdx.x * 64;

    // Stage 64 rows of text into LDS as bf16 (coalesced float4 loads).
    for (int it = 0; it < 32; ++it) {
        const int idx = tid + it*256;             // float4 index in 64x512
        const int r   = idx >> 7;                 // /128
        const int col = (idx & 127) << 2;
        const float4 tv = *(const float4*)(text + (rowbase + r)*DIM + col);
        __bf16* pt = xt + r*DIM + col;
        pt[0] = (__bf16)tv.x; pt[1] = (__bf16)tv.y; pt[2] = (__bf16)tv.z; pt[3] = (__bf16)tv.w;
    }
    __syncthreads();

    const int wave = tid >> 5;
    const int lane = tid & 31;
    const int lrow = lane & 15;
    const int half = lane >> 4;
    const int sub  = wave & 3;     // 16-row subtile
    const int hg   = wave >> 2;    // head-group / ntile-half

    // Hoist all 16 A-fragments (K = 512) for this subtile into registers.
    v16bf afr[16];
    {
        const __bf16* arow = xt + (sub*16 + lrow)*DIM;
#pragma unroll
        for (int kt = 0; kt < 16; ++kt) {
            v8bf a0 = *(const v8bf*)(arow + kt*32 + (half ? 16 : 0));
            v8bf a1 = *(const v8bf*)(arow + kt*32 + (half ? 24 : 8));
#pragma unroll
            for (int e = 0; e < 8; ++e) { afr[kt][e] = a0[e]; afr[kt][8+e] = a1[e]; }
        }
    }

    // Base pointer for this wave's Xa epilogue reads (f32, coalesced).
    const float* xap = acoustic + (rowbase + sub*16 + half*8)*DIM + lrow;

    // ---------------- S part ----------------
    for (int hh = 0; hh < 4; ++hh) {
        const int h = hg*4 + hh;
        float ps[8];
#pragma unroll
        for (int v = 0; v < 8; ++v) ps[v] = 0.f;

        for (int nt = 0; nt < 32; ++nt) {
            // C init = w[h][col]: epilogue multiply by Xa turns this into the
            // (Wa^T bt).x_a bias term summed over columns.
            const float cw = wvec[h*DIM + nt*16 + lrow];
            v8f acc;
#pragma unroll
            for (int v = 0; v < 8; ++v) acc[v] = cw;

            const __bf16* bp = Bswz + ((size_t)((h*32 + nt)*16) << 9) + (lane << 4);
            // Warm L2->L0 for the next 16 KB B block of this head.
            __builtin_prefetch(bp + (16 << 9), 0, 1);
#pragma unroll
            for (int kt = 0; kt < 16; ++kt) {
                v16bf bfrag = *(const v16bf*)(bp + (kt << 9));
                acc = __builtin_amdgcn_wmma_f32_16x16x32_bf16(false, afr[kt], false, bfrag,
                                                              (short)0, acc, false, false);
            }
            // Epilogue: ps[row] += Y[row][col] * Xa[row][col]  (Xa in f32)
#pragma unroll
            for (int v = 0; v < 8; ++v)
                ps[v] += acc[v] * xap[v*DIM + nt*16];
        }
        // Reduce partial sums across each 16-lane half (columns of the tile).
#pragma unroll
        for (int v = 0; v < 8; ++v) {
            ps[v] += __shfl_xor(ps[v], 1, 32);
            ps[v] += __shfl_xor(ps[v], 2, 32);
            ps[v] += __shfl_xor(ps[v], 4, 32);
            ps[v] += __shfl_xor(ps[v], 8, 32);
        }
        if (lrow == 0) {
#pragma unroll
            for (int v = 0; v < 8; ++v) {
                const long r = rowbase + sub*16 + half*8 + v;
                out[r*DIM + h] = ps[v]*0.125f + tb[r*NHEAD + h];
            }
        }
    }

    // ---------------- v_td part ----------------
    for (int ntl = 0; ntl < 16; ++ntl) {
        const int nt  = hg*16 + ntl;
        const int col = nt*16 + lrow;
        const float bv = (col < TDROWS) ? btd[col] : 0.f;
        v8f acc;
#pragma unroll
        for (int v = 0; v < 8; ++v) acc[v] = bv;

        const __bf16* bp = Wtdswz + ((size_t)(nt*16) << 9) + (lane << 4);
        __builtin_prefetch(bp + (16 << 9), 0, 1);
#pragma unroll
        for (int kt = 0; kt < 16; ++kt) {
            v16bf bfrag = *(const v16bf*)(bp + (kt << 9));
            acc = __builtin_amdgcn_wmma_f32_16x16x32_bf16(false, afr[kt], false, bfrag,
                                                          (short)0, acc, false, false);
        }
        if (col < TDROWS) {
#pragma unroll
            for (int v = 0; v < 8; ++v) {
                const long r = rowbase + sub*16 + half*8 + v;
                out[r*DIM + 8 + col] = acc[v];
            }
        }
    }
}

// ===========================================================================
// Host launcher
// Workspace layout (bytes):
//   [0, 4MB)            Bswz   (bf16, 8*512*512)
//   [4MB, 4.5MB)        Wtdswz (bf16, 512*512 padded)
//   [+0, +16KB)         u      (f32, 8*512)
//   [+16KB, +32KB)      w      (f32, 8*512)
//   [+32KB, +32KB+64)   c      (f32, 8, padded)
//   [..., +512KB)       tb     (f32, 16384*8)
// Total ~5.3 MB.
// ===========================================================================
extern "C" void kernel_launch(void* const* d_in, const int* in_sizes, int n_in,
                              void* d_out, int out_size, void* d_ws, size_t ws_size,
                              hipStream_t stream) {
    const float* acoustic = (const float*)d_in[0];
    const float* text     = (const float*)d_in[1];
    const float* Wa       = (const float*)d_in[2];
    const float* ba       = (const float*)d_in[3];
    const float* Wt       = (const float*)d_in[4];
    const float* bt       = (const float*)d_in[5];
    const float* Wtd      = (const float*)d_in[6];
    const float* btd      = (const float*)d_in[7];
    float* out = (float*)d_out;

    char* ws = (char*)d_ws;
    __bf16* Bswz   = (__bf16*)ws;                                   // 4 MB
    __bf16* Wtdswz = (__bf16*)(ws + (4u << 20));                    // 512 KB
    float*  u      = (float*)(ws + (4u << 20) + (512u << 10));
    float*  w      = u + NHEAD*DIM;
    float*  c      = w + NHEAD*DIM;
    float*  tb     = c + 16;                                        // padded

    k_prep_uwc<<<NHEAD, DIM, 0, stream>>>(Wa, ba, Wt, bt, u, w, c);
    k_prep_M  <<<512, 256, 0, stream>>>(Wa, Wt, Bswz);
    k_prep_wtd<<<64, 256, 0, stream>>>(Wtd, Wtdswz);
    k_prep_tb <<<NROWS/8, 256, 0, stream>>>(text, u, c, tb);

    // 64 rows per block -> 256 blocks; 64 KB dynamic LDS (Xt in bf16).
    k_main<<<NROWS/64, 256, 64*DIM*sizeof(__bf16), stream>>>(
        acoustic, text, Bswz, Wtdswz, w, btd, tb, out);
}
